// Attention_31636729103164
// MI455X (gfx1250) — matrix operand order repaired and checked by reading, not compiled, not run on hardware
//
#include <hip/hip_runtime.h>

// ---------------------------------------------------------------------------
// Causal single-head attention, MI455X (gfx1250, wave32, WMMA).
// Phase 1: Q = X@Wq+bq (f16), Kt = (X@Wk+bk)^T (f16), V = X@Wv+bv (f16)
// Phase 2: flash attention, kv tiles of 64, f32 accumulators, f32 out.
// ---------------------------------------------------------------------------

typedef __attribute__((ext_vector_type(16))) _Float16 v16h;
typedef __attribute__((ext_vector_type(8)))  _Float16 v8h;
typedef __attribute__((ext_vector_type(8)))  float    v8f;
typedef __attribute__((ext_vector_type(4)))  float    v4f;

#define SEQ_LEN 2048
#define D_MODEL 1024
#define D_KEY   64
#define BATCH   8
#define MTOT    (BATCH * SEQ_LEN)   // 16384 rows total

// -------- A-fragment loader: 16x32 f16 A tile from a row-major matrix -------
// Layout (ISA 7.12.2, 16-bit A 16x32): lane holds row M=lane%16;
// elements 0..7 = K[base..base+7], 8..15 = K[base+16..base+23],
// base = kb + (lane>=16 ? 8 : 0).  Two 16-byte vector loads.
__device__ inline v16h load_a_frag(const _Float16* __restrict__ rowptr, int base) {
  v8h lo = *(const v8h*)(rowptr + base);
  v8h hh = *(const v8h*)(rowptr + base + 16);
  v16h a;
#pragma unroll
  for (int i = 0; i < 8; ++i) { a[i] = lo[i]; a[i + 8] = hh[i]; }
  return a;
}

// ---------------------------------------------------------------------------
// Phase 1: each wave computes NT consecutive 16x16 tiles (A-frag reuse).
// X:[MTOT,1024] f32, W:[1024,N] f32 -> f16 out (+bias).
// TRANS=true stores out[n][m] (leading dim MTOT) for the K matrix.
// ---------------------------------------------------------------------------
template <bool TRANS, int NT>
__global__ __launch_bounds__(128) void proj_gemm(const float* __restrict__ X,
                                                 const float* __restrict__ W,
                                                 const float* __restrict__ bias,
                                                 _Float16* __restrict__ out,
                                                 int N) {
  const int lane  = threadIdx.x & 31;
  const int wave  = threadIdx.x >> 5;
  const int m0    = blockIdx.x * 16;
  const int n0    = (blockIdx.y * 4 + wave) * (16 * NT);
  const int hi8   = (lane >= 16) ? 8 : 0;
  const int nlane = lane & 15;

  v8f c[NT] = {};
  const float* xrow = X + (size_t)(m0 + nlane) * D_MODEL;

  for (int kb = 0; kb < D_MODEL; kb += 32) {
    // A fragment (shared across the NT output tiles): f32 -> f16 convert.
    v16h a;
    {
      const float* p = xrow + kb + hi8;
      v4f x0 = *(const v4f*)(p);
      v4f x1 = *(const v4f*)(p + 4);
      v4f x2 = *(const v4f*)(p + 16);
      v4f x3 = *(const v4f*)(p + 20);
#pragma unroll
      for (int i = 0; i < 4; ++i) {
        a[i]      = (_Float16)x0[i];
        a[i + 4]  = (_Float16)x1[i];
        a[i + 8]  = (_Float16)x2[i];
        a[i + 12] = (_Float16)x3[i];
      }
    }
#pragma unroll
    for (int t = 0; t < NT; ++t) {
      // B fragment: lane = column n0+16t+nlane, elements = K rows kb+hi8+{0..7,16..23}.
      v16h b;
      const float* p = W + (size_t)(kb + hi8) * N + n0 + 16 * t + nlane;
#pragma unroll
      for (int i = 0; i < 8; ++i) {
        b[i]     = (_Float16)p[(size_t)i * N];
        b[i + 8] = (_Float16)p[(size_t)(i + 16) * N];
      }
      c[t] = __builtin_amdgcn_wmma_f32_16x16x32_f16(false, a, false, b,
                                                    (short)0, c[t], false, false);
    }
  }

#pragma unroll
  for (int t = 0; t < NT; ++t) {
    const float bv = bias[n0 + 16 * t + nlane];
#pragma unroll
    for (int r = 0; r < 8; ++r) {
      const int   row = m0 + r + hi8;
      const float val = c[t][r] + bv;
      if (TRANS)
        out[(size_t)(n0 + 16 * t + nlane) * MTOT + row] = (_Float16)val;
      else
        out[(size_t)row * N + n0 + 16 * t + nlane] = (_Float16)val;
    }
  }
}

// ---------------------------------------------------------------------------
// Phase 2: flash attention.
// Block = 256 threads = 8 waves; block handles 16 query rows of one batch.
// Each wave owns a 128-wide slice of the d_model=1024 output columns.
// kv tiles of 64; per step/wave: 8 score WMMAs + 16 PV WMMAs, so the
// softmax reductions / exp / accumulator rescale amortize over 2x the
// matrix work vs a 32-wide tile (inner loop was VALU-issue bound).
// ---------------------------------------------------------------------------
__global__ __launch_bounds__(256) void attn_kernel(
    const _Float16* __restrict__ Qb,   // [MTOT, 64]   row-major
    const _Float16* __restrict__ Kt,   // [64, MTOT]   (K transposed)
    const _Float16* __restrict__ Vb,   // [MTOT, 1024] row-major
    float* __restrict__ out) {         // [MTOT, 1024] f32
  __shared__ _Float16 pbuf[8][16 * 64];   // per-wave P staging (C -> A layout)

  const int    lane    = threadIdx.x & 31;
  const int    wave    = threadIdx.x >> 5;
  const int    qbase   = blockIdx.x * 16;              // seq position in batch
  const size_t rowbase = (size_t)blockIdx.y * SEQ_LEN; // global row base
  const int    hi8     = (lane >= 16) ? 8 : 0;
  const int    nlane   = lane & 15;
  const int    c0      = wave * 128;                   // output column slice

  // Q A-fragments (d_k = 64 -> two 16x32 fragments), loaded once.
  const _Float16* qrow = Qb + (rowbase + qbase + nlane) * D_KEY;
  const v16h qa0 = load_a_frag(qrow, hi8);
  const v16h qa1 = load_a_frag(qrow, 32 + hi8);

  v8f acc[8];
#pragma unroll
  for (int t = 0; t < 8; ++t) acc[t] = (v8f){};
  float mrow[8], lrow[8];
#pragma unroll
  for (int r = 0; r < 8; ++r) { mrow[r] = -1e30f; lrow[r] = 0.0f; }

  const float sc = 0.125f * 1.44269504088896f;  // 1/sqrt(64) * log2(e)

  const int kv_end = qbase + 16;                 // causal bound (exclusive)
  for (int kvb = 0; kvb < kv_end; kvb += 64) {
    // prefetch next kv step's V rows into cache (global_prefetch_b8 path)
    if (kvb + 64 < kv_end)
      __builtin_prefetch(Vb + (size_t)(rowbase + kvb + 64 + lane) * D_MODEL + c0,
                         0, 1);

    // ---- scores: S[16x64] = Q[16x64] @ K^T slice -------------------------
    v8f s[4] = {};
#pragma unroll
    for (int kb = 0; kb < D_KEY; kb += 32) {
      const _Float16* kp = Kt + (size_t)(kb + hi8) * MTOT + rowbase + kvb + nlane;
      const v16h aq = (kb == 0) ? qa0 : qa1;
#pragma unroll
      for (int tt = 0; tt < 4; ++tt) {
        v16h kbf;
#pragma unroll
        for (int i = 0; i < 8; ++i) {
          kbf[i]     = kp[16 * tt + (size_t)i * MTOT];
          kbf[i + 8] = kp[16 * tt + (size_t)(i + 16) * MTOT];
        }
        s[tt] = __builtin_amdgcn_wmma_f32_16x16x32_f16(false, aq, false, kbf,
                                                       (short)0, s[tt], false,
                                                       false);
      }
    }
#pragma unroll
    for (int tt = 0; tt < 4; ++tt)
#pragma unroll
      for (int r = 0; r < 8; ++r) s[tt][r] *= sc;

    // causal masking only needed on tail tiles (uniform per wave)
    if (kvb + 63 > qbase) {
#pragma unroll
      for (int tt = 0; tt < 4; ++tt)
#pragma unroll
        for (int r = 0; r < 8; ++r) {
          const int q = qbase + r + hi8;
          if (kvb + 16 * tt + nlane > q) s[tt][r] = -1e30f;
        }
    }

    // ---- online softmax (exp2 domain) ------------------------------------
    float pscale[8];
#pragma unroll
    for (int r = 0; r < 8; ++r) {
      float mx = fmaxf(fmaxf(s[0][r], s[1][r]), fmaxf(s[2][r], s[3][r]));
#pragma unroll
      for (int off = 8; off >= 1; off >>= 1)
        mx = fmaxf(mx, __shfl_xor(mx, off, 32));
      const float mnew = fmaxf(mrow[r], mx);
      const float sco  = exp2f(mrow[r] - mnew);
      float p[4], ps = 0.0f;
#pragma unroll
      for (int tt = 0; tt < 4; ++tt) { p[tt] = exp2f(s[tt][r] - mnew); ps += p[tt]; }
#pragma unroll
      for (int off = 8; off >= 1; off >>= 1)
        ps += __shfl_xor(ps, off, 32);
      lrow[r]   = lrow[r] * sco + ps;
      mrow[r]   = mnew;
      pscale[r] = sco;
      // stash P (f16) row-major [16][64]; C layout: row r+hi8, col 16*tt+nlane
#pragma unroll
      for (int tt = 0; tt < 4; ++tt)
        pbuf[wave][(r + hi8) * 64 + 16 * tt + nlane] = (_Float16)p[tt];
    }

    // rescale running accumulators
#pragma unroll
    for (int t = 0; t < 8; ++t)
#pragma unroll
      for (int r = 0; r < 8; ++r) acc[t][r] *= pscale[r];

    // P as two A-fragments (from LDS; same-wave DS ops are in-order)
    const _Float16* prow = &pbuf[wave][nlane * 64];
    const v16h pa0 = load_a_frag(prow, hi8);
    const v16h pa1 = load_a_frag(prow, 32 + hi8);

    // ---- acc += P[16x64] @ V[64x128 slice] --------------------------------
#pragma unroll
    for (int t = 0; t < 8; ++t) {
      const _Float16* vp =
          Vb + (size_t)(rowbase + kvb + hi8) * D_MODEL + c0 + 16 * t + nlane;
      v16h vb0, vb1;
#pragma unroll
      for (int i = 0; i < 8; ++i) {
        vb0[i]     = vp[(size_t)i * D_MODEL];
        vb0[i + 8] = vp[(size_t)(i + 16) * D_MODEL];
        vb1[i]     = vp[(size_t)(i + 32) * D_MODEL];
        vb1[i + 8] = vp[(size_t)(i + 48) * D_MODEL];
      }
      acc[t] = __builtin_amdgcn_wmma_f32_16x16x32_f16(false, pa0, false, vb0,
                                                      (short)0, acc[t], false,
                                                      false);
      acc[t] = __builtin_amdgcn_wmma_f32_16x16x32_f16(false, pa1, false, vb1,
                                                      (short)0, acc[t], false,
                                                      false);
    }
  }

  // ---- epilogue: normalize and store f32 ----------------------------------
#pragma unroll
  for (int r = 0; r < 8; ++r) {
    const float  inv  = 1.0f / lrow[r];
    const size_t orow = (rowbase + qbase + r + hi8) * (size_t)D_MODEL;
#pragma unroll
    for (int t = 0; t < 8; ++t)
      out[orow + c0 + 16 * t + nlane] = acc[t][r] * inv;
  }
}

// ---------------------------------------------------------------------------
extern "C" void kernel_launch(void* const* d_in, const int* in_sizes, int n_in,
                              void* d_out, int out_size, void* d_ws,
                              size_t ws_size, hipStream_t stream) {
  const float* X  = (const float*)d_in[0];
  const float* Wq = (const float*)d_in[1];
  const float* bq = (const float*)d_in[2];
  const float* Wk = (const float*)d_in[3];
  const float* bk = (const float*)d_in[4];
  const float* Wv = (const float*)d_in[5];
  const float* bv = (const float*)d_in[6];
  float* out = (float*)d_out;

  // workspace: Q (2 MB) | K^T (2 MB) | V (32 MB), all f16
  _Float16* Qb = (_Float16*)d_ws;
  _Float16* Kt = Qb + (size_t)MTOT * D_KEY;
  _Float16* Vb = Kt + (size_t)MTOT * D_KEY;

  dim3 blk(128);
  // Q, K: N=64 -> one 16-wide tile per wave, K stored transposed.
  proj_gemm<false, 1><<<dim3(MTOT / 16, 1), blk, 0, stream>>>(X, Wq, bq, Qb, D_KEY);
  proj_gemm<true,  1><<<dim3(MTOT / 16, 1), blk, 0, stream>>>(X, Wk, bk, Kt, D_KEY);
  // V: N=1024 -> 4 tiles per wave (A-fragment reused 4x), grid.y = 1024/256.
  proj_gemm<false, 4><<<dim3(MTOT / 16, D_MODEL / 256), blk, 0, stream>>>(X, Wv, bv, Vb, D_MODEL);

  attn_kernel<<<dim3(SEQ_LEN / 16, BATCH), dim3(256), 0, stream>>>(Qb, Kt, Vb, out);
}
